// GMT_85358180041227
// MI455X (gfx1250) — compile-verified, matrix operand order checked by reading, not run on hardware
//
#include <hip/hip_runtime.h>
#include <hip/hip_bf16.h>
#include <math.h>

// Problem constants (match reference)
#define NNODES 20000
#define NEDGES 640000
#define FEAT   128
#define NGRAPH 128
#define NHEAD  4
#define CH     64
#define HC     256           // NHEAD*CH
#define NCLS   10

typedef __attribute__((ext_vector_type(4)))  __bf16 v4bf;
typedef __attribute__((ext_vector_type(8)))  __bf16 v8bf;
typedef __attribute__((ext_vector_type(16))) __bf16 v16bf;
typedef __attribute__((ext_vector_type(8)))  float  v8f;

// ---------------------------------------------------------------------------
// generic fill / relu
// ---------------------------------------------------------------------------
__global__ void fill_kernel(float* p, float v, int n) {
  int i = blockIdx.x * blockDim.x + threadIdx.x;
  if (i < n) p[i] = v;
}

__global__ void relu_kernel(float* p, int n) {
  int i = blockIdx.x * blockDim.x + threadIdx.x;
  if (i < n) p[i] = fmaxf(p[i], 0.0f);
}

// fp32 -> bf16 (vectorized, n4 = n/4 quads)
__global__ void cvt_bf16_kernel(const float* __restrict__ src,
                                __bf16* __restrict__ dst, int n4) {
  int i = blockIdx.x * blockDim.x + threadIdx.x;
  if (i >= n4) return;
  float4 f = ((const float4*)src)[i];
  v4bf h;
  h[0] = (__bf16)f.x; h[1] = (__bf16)f.y;
  h[2] = (__bf16)f.z; h[3] = (__bf16)f.w;
  ((v4bf*)dst)[i] = h;
}

// fp32 W[K,N] -> bf16 Wt[N,K] (transpose, done once per weight)
__global__ void cvt_w_kernel(const float* __restrict__ W,
                             __bf16* __restrict__ Wt, int K, int N) {
  int i = blockIdx.x * blockDim.x + threadIdx.x;
  if (i >= K * N) return;
  int k = i / N, n = i % N;
  Wt[(size_t)n * K + k] = (__bf16)W[i];
}

// ---------------------------------------------------------------------------
// WMMA bf16 GEMM:  Out[M,Nc] = A[M,K] @ Wt[Nc,K]^T + bias
// A, Wt pre-converted bf16 (Wt column-major), fp32 accumulate/output.
// Block = 256 threads (8 waves).  Tile = 128 rows x 64 cols, K step 32.
// Tiles staged with GLOBAL_LOAD_ASYNC_TO_LDS_B128 (ASYNCcnt), fragments read
// as 16 B-aligned ds_load_b128 pairs; one A fragment feeds 4 v_wmma per step.
// ---------------------------------------------------------------------------
#define TM    128
#define TN    64
#define KSTEP 32
#define LDA   40   // padded stride: rows are 80 B (16 B aligned), pad kills conflicts
#define LDB   40

__device__ __forceinline__ unsigned lds_addr32(const void* p) {
  // ISA 10.2: for LDS apertures, LDS byte address = flat addr[31:0]
  return (unsigned)(size_t)p;
}

template <int K>
__global__ __launch_bounds__(256)
void gemm_bias_wmma(const __bf16* __restrict__ A, const __bf16* __restrict__ Wt,
                    const float* __restrict__ bias, float* __restrict__ Out,
                    int M, int Nc)
{
  __shared__ __align__(16) __bf16 As[TM * LDA];   // 128 rows x 32 K (padded)
  __shared__ __align__(16) __bf16 Bs[TN * LDB];   // 64 cols x 32 K (padded)

  const int tid  = threadIdx.x;
  const int wave = tid >> 5;
  const int lane = tid & 31;
  const int rowTile = blockIdx.x * TM;
  const int colTile = blockIdx.y * TN;

  v8f acc0 = {}, acc1 = {}, acc2 = {}, acc3 = {};

  const int mrow = (wave << 4) + (lane & 15);
  const int ncol = lane & 15;
  const int kb   = (lane >> 4) << 3;   // 0 or 8

#pragma unroll
  for (int k0 = 0; k0 < K; k0 += KSTEP) {
    // ---- async copy A tile: 128x32 bf16 = 512 x 16 B chunks (2/thread)
#pragma unroll
    for (int i = tid; i < TM * KSTEP / 8; i += 256) {
      int r  = i >> 2;             // 4 chunks per row
      int c8 = (i & 3) << 3;
      int gr = rowTile + r; if (gr >= M) gr = M - 1;   // clamp: masked at store
      const __bf16* gsrc = A + (size_t)gr * K + k0 + c8;
      unsigned lds = lds_addr32(&As[r * LDA + c8]);
      asm volatile("global_load_async_to_lds_b128 %0, %1, off"
                   :: "v"(lds), "v"((unsigned long long)(size_t)gsrc)
                   : "memory");
    }
    // ---- async copy W tile: 64 cols x 32 K bf16 = 256 x 16 B chunks (1/thread)
    {
      int c  = tid >> 2;
      int k8 = (tid & 3) << 3;
      const __bf16* gsrc = Wt + (size_t)(colTile + c) * K + k0 + k8;
      unsigned lds = lds_addr32(&Bs[c * LDB + k8]);
      asm volatile("global_load_async_to_lds_b128 %0, %1, off"
                   :: "v"(lds), "v"((unsigned long long)(size_t)gsrc)
                   : "memory");
    }
    asm volatile("s_wait_asynccnt 0x0" ::: "memory");
    __syncthreads();

    // ---- A fragment: two contiguous 16 B runs per lane (ISA 16-bit layout)
    const __bf16* ap = &As[mrow * LDA];
    v8bf alo = *(const v8bf*)(ap + kb);
    v8bf ahi = *(const v8bf*)(ap + 16 + kb);
    v16bf afrag = __builtin_shufflevector(alo, ahi,
        0, 1, 2, 3, 4, 5, 6, 7, 8, 9, 10, 11, 12, 13, 14, 15);

    // ---- 4 column sub-tiles, one WMMA each, A fragment reused
#pragma unroll
    for (int t = 0; t < 4; ++t) {
      const __bf16* bp = &Bs[(t * 16 + ncol) * LDB];
      v8bf blo = *(const v8bf*)(bp + kb);
      v8bf bhi = *(const v8bf*)(bp + 16 + kb);
      v16bf bfrag = __builtin_shufflevector(blo, bhi,
          0, 1, 2, 3, 4, 5, 6, 7, 8, 9, 10, 11, 12, 13, 14, 15);
      if (t == 0)
        acc0 = __builtin_amdgcn_wmma_f32_16x16x32_bf16(false, afrag, false, bfrag,
                                                       (short)0, acc0, false, false);
      else if (t == 1)
        acc1 = __builtin_amdgcn_wmma_f32_16x16x32_bf16(false, afrag, false, bfrag,
                                                       (short)0, acc1, false, false);
      else if (t == 2)
        acc2 = __builtin_amdgcn_wmma_f32_16x16x32_bf16(false, afrag, false, bfrag,
                                                       (short)0, acc2, false, false);
      else
        acc3 = __builtin_amdgcn_wmma_f32_16x16x32_bf16(false, afrag, false, bfrag,
                                                       (short)0, acc3, false, false);
    }
    __syncthreads();
  }

  // ---- store: VGPR r -> row r (lanes 0-15) / r+8 (lanes 16-31), col = lane&15
#pragma unroll
  for (int t = 0; t < 4; ++t) {
    const v8f& acc = (t == 0) ? acc0 : (t == 1) ? acc1 : (t == 2) ? acc2 : acc3;
    int col = colTile + t * 16 + (lane & 15);
    float b = bias[col];
#pragma unroll
    for (int r = 0; r < 8; ++r) {
      int row = rowTile + (wave << 4) + r + ((lane >> 4) << 3);
      if (row < M) Out[(size_t)row * Nc + col] = acc[r] + b;
    }
  }
}

// ---------------------------------------------------------------------------
// float atomic max via order-preserving int trick
// ---------------------------------------------------------------------------
__device__ __forceinline__ void atomicMaxF(float* addr, float val) {
  if (val >= 0.0f) atomicMax((int*)addr, __float_as_int(val));
  else             atomicMin((unsigned int*)addr, (unsigned int)__float_as_int(val));
}

// ---------------------------------------------------------------------------
// per-edge attention logits  alpha[e,h] = <q[dst,h,:], k[src,h,:]>/sqrt(C)
// and running segment max over dst.  One thread per (edge, head).
// ---------------------------------------------------------------------------
__global__ void edge_logits_kernel(const float* __restrict__ q,
                                   const float* __restrict__ kmat,
                                   const int* __restrict__ src,
                                   const int* __restrict__ dst,
                                   float* __restrict__ alpha,
                                   float* __restrict__ segmax)
{
  int idx = blockIdx.x * blockDim.x + threadIdx.x;
  if (idx >= NEDGES * NHEAD) return;
  int e = idx >> 2, h = idx & 3;
  int s = src[e], d = dst[e];
  const float4* qp = (const float4*)&q[(size_t)d * HC + h * CH];
  const float4* kp = (const float4*)&kmat[(size_t)s * HC + h * CH];
  float acc = 0.0f;
#pragma unroll
  for (int i = 0; i < CH / 4; ++i) {
    float4 a = qp[i], b = kp[i];
    acc += a.x * b.x + a.y * b.y + a.z * b.z + a.w * b.w;
  }
  acc *= 0.125f;   // 1/sqrt(64)
  alpha[idx] = acc;
  atomicMaxF(&segmax[d * NHEAD + h], acc);
}

// p = exp(alpha - max[dst]);  segsum[dst] += p
__global__ void edge_exp_kernel(const float* __restrict__ alpha,
                                const int* __restrict__ dst,
                                const float* __restrict__ segmax,
                                float* __restrict__ p,
                                float* __restrict__ segsum)
{
  int idx = blockIdx.x * blockDim.x + threadIdx.x;
  if (idx >= NEDGES * NHEAD) return;
  int e = idx >> 2, h = idx & 3;
  int d = dst[e];
  float pe = __expf(alpha[idx] - segmax[d * NHEAD + h]);
  p[idx] = pe;
  atomicAdd(&segsum[d * NHEAD + h], pe);
}

// out[dst,h,:] += (p / (sum[dst,h]+eps)) * v[src,h,:]
__global__ void edge_scatter_kernel(const float* __restrict__ p,
                                    const float* __restrict__ segsum,
                                    const float* __restrict__ vmat,
                                    const int* __restrict__ src,
                                    const int* __restrict__ dst,
                                    float* __restrict__ out)
{
  int idx = blockIdx.x * blockDim.x + threadIdx.x;
  if (idx >= NEDGES * NHEAD) return;
  int e = idx >> 2, h = idx & 3;
  int s = src[e], d = dst[e];
  float coef = p[idx] / (segsum[d * NHEAD + h] + 1e-16f);
  const float4* vp = (const float4*)&vmat[(size_t)s * HC + h * CH];
  float* op = &out[(size_t)d * HC + h * CH];
#pragma unroll
  for (int i = 0; i < CH / 4; ++i) {
    float4 a = vp[i];
    atomicAdd(op + 4 * i + 0, coef * a.x);
    atomicAdd(op + 4 * i + 1, coef * a.y);
    atomicAdd(op + 4 * i + 2, coef * a.z);
    atomicAdd(op + 4 * i + 3, coef * a.w);
  }
}

// ---------------------------------------------------------------------------
// global mean pool accumulation
// ---------------------------------------------------------------------------
__global__ void pool_acc_kernel(const float* __restrict__ h,
                                const int* __restrict__ batch,
                                float* __restrict__ pooled)
{
  int idx = blockIdx.x * blockDim.x + threadIdx.x;
  if (idx >= NNODES * HC) return;
  int n = idx >> 8;   // /256
  int c = idx & 255;
  atomicAdd(&pooled[batch[n] * HC + c], h[idx]);
}

__global__ void pool_cnt_kernel(const int* __restrict__ batch, float* __restrict__ cnt)
{
  int n = blockIdx.x * blockDim.x + threadIdx.x;
  if (n < NNODES) atomicAdd(&cnt[batch[n]], 1.0f);
}

// ---------------------------------------------------------------------------
// head: mean -> fc1(+relu) -> fc2 -> log_softmax.  One block (64 thr) / graph.
// ---------------------------------------------------------------------------
__global__ __launch_bounds__(64)
void head_kernel(const float* __restrict__ pooled, const float* __restrict__ cnt,
                 const float* __restrict__ Wfc1, const float* __restrict__ bfc1,
                 const float* __restrict__ Wfc2, const float* __restrict__ bfc2,
                 float* __restrict__ out)
{
  __shared__ float hv[CH];
  __shared__ float lg[NCLS];
  int g = blockIdx.x;
  int t = threadIdx.x;
  float inv = 1.0f / fmaxf(cnt[g], 1.0f);
  float acc = bfc1[t];
  for (int i = 0; i < HC; ++i)
    acc += pooled[g * HC + i] * inv * Wfc1[i * CH + t];
  hv[t] = fmaxf(acc, 0.0f);
  __syncthreads();
  if (t < NCLS) {
    float a = bfc2[t];
    for (int i = 0; i < CH; ++i) a += hv[i] * Wfc2[i * NCLS + t];
    lg[t] = a;
  }
  __syncthreads();
  if (t == 0) {
    float m = lg[0];
    for (int j = 1; j < NCLS; ++j) m = fmaxf(m, lg[j]);
    float ssum = 0.0f;
    for (int j = 0; j < NCLS; ++j) ssum += __expf(lg[j] - m);
    float lse = m + __logf(ssum);
    for (int j = 0; j < NCLS; ++j) out[g * NCLS + j] = lg[j] - lse;
  }
}

// ---------------------------------------------------------------------------
// orchestration
// ---------------------------------------------------------------------------
extern "C" void kernel_launch(void* const* d_in, const int* in_sizes, int n_in,
                              void* d_out, int out_size, void* d_ws, size_t ws_size,
                              hipStream_t stream)
{
  (void)in_sizes; (void)n_in; (void)out_size; (void)ws_size;

  const float* x    = (const float*)d_in[0];
  const int*   ei   = (const int*)d_in[1];
  const int*   src  = ei;
  const int*   dst  = ei + NEDGES;
  const int*   batch = (const int*)d_in[2];
  const float *Wq1 = (const float*)d_in[3],  *bq1 = (const float*)d_in[4];
  const float *Wk1 = (const float*)d_in[5],  *bk1 = (const float*)d_in[6];
  const float *Wv1 = (const float*)d_in[7],  *bv1 = (const float*)d_in[8];
  const float *Ws1 = (const float*)d_in[9],  *bs1 = (const float*)d_in[10];
  const float *Wq2 = (const float*)d_in[11], *bq2 = (const float*)d_in[12];
  const float *Wk2 = (const float*)d_in[13], *bk2 = (const float*)d_in[14];
  const float *Wv2 = (const float*)d_in[15], *bv2 = (const float*)d_in[16];
  const float *Ws2 = (const float*)d_in[17], *bs2 = (const float*)d_in[18];
  const float *Wfc1 = (const float*)d_in[19], *bfc1 = (const float*)d_in[20];
  const float *Wfc2 = (const float*)d_in[21], *bfc2 = (const float*)d_in[22];

  // workspace carve-up: fp32 region first, then bf16 region (16 B aligned)
  float* ws = (float*)d_ws;
  size_t o = 0;
  float* q     = ws + o; o += (size_t)NNODES * HC;
  float* k     = ws + o; o += (size_t)NNODES * HC;
  float* v     = ws + o; o += (size_t)NNODES * HC;
  float* h1    = ws + o; o += (size_t)NNODES * HC;
  float* h2    = ws + o; o += (size_t)NNODES * HC;
  float* alpha = ws + o; o += (size_t)NEDGES * NHEAD;
  float* pexp  = ws + o; o += (size_t)NEDGES * NHEAD;
  float* segm  = ws + o; o += (size_t)NNODES * NHEAD;
  float* segs  = ws + o; o += (size_t)NNODES * NHEAD;
  float* pooled= ws + o; o += (size_t)NGRAPH * HC;
  float* cnt   = ws + o; o += (size_t)NGRAPH;

  __bf16* bfb = (__bf16*)(ws + o);
  size_t ob = 0;
  __bf16* xb   = bfb + ob; ob += (size_t)NNODES * FEAT;   // bf16 x
  __bf16* hb   = bfb + ob; ob += (size_t)NNODES * HC;     // bf16 h1
  __bf16* Wq1t = bfb + ob; ob += (size_t)HC * FEAT;
  __bf16* Wk1t = bfb + ob; ob += (size_t)HC * FEAT;
  __bf16* Wv1t = bfb + ob; ob += (size_t)HC * FEAT;
  __bf16* Ws1t = bfb + ob; ob += (size_t)HC * FEAT;
  __bf16* Wq2t = bfb + ob; ob += (size_t)HC * HC;
  __bf16* Wk2t = bfb + ob; ob += (size_t)HC * HC;
  __bf16* Wv2t = bfb + ob; ob += (size_t)HC * HC;
  __bf16* Ws2t = bfb + ob; ob += (size_t)HC * HC;

  dim3 gblk(256);
  dim3 ggrd((NNODES + TM - 1) / TM, HC / TN);
  const int EH  = NEDGES * NHEAD;
  const int NH  = NNODES * NHEAD;
  const int NHC = NNODES * HC;
  dim3 b256(256);
  dim3 gEH((EH + 255) / 256);
  dim3 gNH((NH + 255) / 256);
  dim3 gNHC((NHC + 255) / 256);
  const int W1E = FEAT * HC, W2E = HC * HC;
  dim3 gW1((W1E + 255) / 256), gW2((W2E + 255) / 256);

  // ---------- one-time bf16 conversion / weight transpose ----------
  cvt_bf16_kernel<<<((NNODES * FEAT / 4) + 255) / 256, b256, 0, stream>>>(x, xb, NNODES * FEAT / 4);
  cvt_w_kernel<<<gW1, b256, 0, stream>>>(Wq1, Wq1t, FEAT, HC);
  cvt_w_kernel<<<gW1, b256, 0, stream>>>(Wk1, Wk1t, FEAT, HC);
  cvt_w_kernel<<<gW1, b256, 0, stream>>>(Wv1, Wv1t, FEAT, HC);
  cvt_w_kernel<<<gW1, b256, 0, stream>>>(Ws1, Ws1t, FEAT, HC);
  cvt_w_kernel<<<gW2, b256, 0, stream>>>(Wq2, Wq2t, HC, HC);
  cvt_w_kernel<<<gW2, b256, 0, stream>>>(Wk2, Wk2t, HC, HC);
  cvt_w_kernel<<<gW2, b256, 0, stream>>>(Wv2, Wv2t, HC, HC);
  cvt_w_kernel<<<gW2, b256, 0, stream>>>(Ws2, Ws2t, HC, HC);

  // ---------- layer 1 ----------
  gemm_bias_wmma<FEAT><<<ggrd, gblk, 0, stream>>>(xb, Wq1t, bq1, q,  NNODES, HC);
  gemm_bias_wmma<FEAT><<<ggrd, gblk, 0, stream>>>(xb, Wk1t, bk1, k,  NNODES, HC);
  gemm_bias_wmma<FEAT><<<ggrd, gblk, 0, stream>>>(xb, Wv1t, bv1, v,  NNODES, HC);
  gemm_bias_wmma<FEAT><<<ggrd, gblk, 0, stream>>>(xb, Ws1t, bs1, h1, NNODES, HC);

  fill_kernel<<<gNH, b256, 0, stream>>>(segm, -1e30f, NH);
  fill_kernel<<<gNH, b256, 0, stream>>>(segs, 0.0f, NH);
  edge_logits_kernel<<<gEH, b256, 0, stream>>>(q, k, src, dst, alpha, segm);
  edge_exp_kernel<<<gEH, b256, 0, stream>>>(alpha, dst, segm, pexp, segs);
  edge_scatter_kernel<<<gEH, b256, 0, stream>>>(pexp, segs, v, src, dst, h1);
  relu_kernel<<<gNHC, b256, 0, stream>>>(h1, NHC);
  cvt_bf16_kernel<<<((NHC / 4) + 255) / 256, b256, 0, stream>>>(h1, hb, NHC / 4);

  // ---------- layer 2 ----------
  gemm_bias_wmma<HC><<<ggrd, gblk, 0, stream>>>(hb, Wq2t, bq2, q,  NNODES, HC);
  gemm_bias_wmma<HC><<<ggrd, gblk, 0, stream>>>(hb, Wk2t, bk2, k,  NNODES, HC);
  gemm_bias_wmma<HC><<<ggrd, gblk, 0, stream>>>(hb, Wv2t, bv2, v,  NNODES, HC);
  gemm_bias_wmma<HC><<<ggrd, gblk, 0, stream>>>(hb, Ws2t, bs2, h2, NNODES, HC);

  fill_kernel<<<gNH, b256, 0, stream>>>(segm, -1e30f, NH);
  fill_kernel<<<gNH, b256, 0, stream>>>(segs, 0.0f, NH);
  edge_logits_kernel<<<gEH, b256, 0, stream>>>(q, k, src, dst, alpha, segm);
  edge_exp_kernel<<<gEH, b256, 0, stream>>>(alpha, dst, segm, pexp, segs);
  edge_scatter_kernel<<<gEH, b256, 0, stream>>>(pexp, segs, v, src, dst, h2);
  relu_kernel<<<gNHC, b256, 0, stream>>>(h2, NHC);

  // ---------- pool + head ----------
  int GHC = NGRAPH * HC;
  fill_kernel<<<(GHC + 255) / 256, b256, 0, stream>>>(pooled, 0.0f, GHC);
  fill_kernel<<<1, 128, 0, stream>>>(cnt, 0.0f, NGRAPH);
  pool_acc_kernel<<<gNHC, b256, 0, stream>>>(h2, batch, pooled);
  pool_cnt_kernel<<<(NNODES + 255) / 256, b256, 0, stream>>>(batch, cnt);
  head_kernel<<<NGRAPH, 64, 0, stream>>>(pooled, cnt, Wfc1, bfc1, Wfc2, bfc2,
                                         (float*)d_out);
}
